// CompressiveMemory_7473243095368
// MI455X (gfx1250) — matrix operand order repaired and checked by reading, not compiled, not run on hardware
//
#include <hip/hip_runtime.h>
#include <cstdint>
#include <cstddef>

#define DEV __device__ __forceinline__

typedef __attribute__((ext_vector_type(16))) __bf16 v16bf;
typedef __attribute__((ext_vector_type(8)))  __bf16 v8bf;
typedef __attribute__((ext_vector_type(8)))  float  v8f;

// Problem sizes (fixed by the reference)
constexpr int Bc    = 4;
constexpr int Sc    = 8192;
constexpr int Dc    = 512;
constexpr int Hc    = 8;
constexpr int DKc   = 64;
constexpr int DVc   = 64;
constexpr int Lc    = 512;
constexpr int NSEGc = Sc / Lc;    // 16
constexpr int BHc   = Bc * Hc;    // 32
constexpr int MROWS = Bc * Sc;    // 32768

constexpr float NEG_BIG = -1e30f;

DEV v8f vzero8() {
  v8f z;
#pragma unroll
  for (int i = 0; i < 8; ++i) z[i] = 0.f;
  return z;
}

DEV v8f wmma_bf16(v16bf a, v16bf b, v8f c) {
  return __builtin_amdgcn_wmma_f32_16x16x32_bf16(
      /*neg_a=*/false, a, /*neg_b=*/false, b,
      /*c_mod=*/(short)0, c, /*reuse_a=*/false, /*reuse_b=*/false);
}

// A-fragment (16M x 32K) from row-major bf16, row stride ld.
// Layout: lane m = lane&15; lanes 0-15 hold K=0..15, lanes 16-31 hold K=16..31 (matches load_fragB)
DEV v16bf load_fragA(const __bf16* p, int ld, int row, int k0, int lane) {
  int m = lane & 15, hb = (lane >> 4) * 16;
  return *(const v16bf*)(p + (size_t)(row + m) * ld + k0 + hb);
}

// B-fragment (32K x 16N); source is B-transposed (N x K) row-major, row stride ld.
// Layout (contiguous k, as in load_fragA): lane n = lane&15; lanes 0-15 hold K=0..15, lanes 16-31 hold K=16..31
DEV v16bf load_fragB(const __bf16* p, int ld, int col, int k0, int lane) {
  int n = lane & 15, hb = (lane >> 4) * 16;
  return *(const v16bf*)(p + (size_t)(col + n) * ld + k0 + hb);
}

DEV float elu1(float x) { return x > 0.f ? x + 1.f : __expf(x); }

// ---------------------------------------------------------------------------
// K0a: fp32 -> bf16 elementwise
__global__ void cvt_f32_bf16(const float* __restrict__ in, __bf16* __restrict__ out, int n) {
  for (int i = blockIdx.x * 256 + threadIdx.x; i < n; i += gridDim.x * 256)
    out[i] = (__bf16)in[i];
}

// K0b: transpose-convert a 512x512 fp32 matrix into bf16 (out[n][k] = in[k][n])
__global__ void cvtT512_f32_bf16(const float* __restrict__ in, __bf16* __restrict__ out) {
  int i = blockIdx.x * 256 + threadIdx.x;   // 262144 total
  int nr = i >> 9, k = i & 511;
  out[i] = (__bf16)in[k * 512 + nr];
}

// ---------------------------------------------------------------------------
// K1: C(bf16, MxN) = A(bf16, MxK) @ Bt(bf16, NxK stored row-major) ; fp32 accum
// Block tile 128x128, 8 waves, wave tile 32x64 (2x4 WMMA tiles)
__global__ void gemm_bf16_obf16(const __bf16* __restrict__ A, const __bf16* __restrict__ Bt,
                                __bf16* __restrict__ C, int M, int N, int K) {
  int lane = threadIdx.x & 31, wave = threadIdx.x >> 5;
  int row = blockIdx.x * 128 + (wave >> 1) * 32;
  int col = blockIdx.y * 128 + (wave & 1) * 64;
  v8f acc[2][4];
#pragma unroll
  for (int i = 0; i < 2; ++i)
#pragma unroll
    for (int j = 0; j < 4; ++j) acc[i][j] = vzero8();

  for (int k0 = 0; k0 < K; k0 += 32) {
    v16bf a0 = load_fragA(A, K, row,      k0, lane);
    v16bf a1 = load_fragA(A, K, row + 16, k0, lane);
    v16bf b0 = load_fragB(Bt, K, col,      k0, lane);
    v16bf b1 = load_fragB(Bt, K, col + 16, k0, lane);
    v16bf b2 = load_fragB(Bt, K, col + 32, k0, lane);
    v16bf b3 = load_fragB(Bt, K, col + 48, k0, lane);
    acc[0][0] = wmma_bf16(a0, b0, acc[0][0]);
    acc[0][1] = wmma_bf16(a0, b1, acc[0][1]);
    acc[0][2] = wmma_bf16(a0, b2, acc[0][2]);
    acc[0][3] = wmma_bf16(a0, b3, acc[0][3]);
    acc[1][0] = wmma_bf16(a1, b0, acc[1][0]);
    acc[1][1] = wmma_bf16(a1, b1, acc[1][1]);
    acc[1][2] = wmma_bf16(a1, b2, acc[1][2]);
    acc[1][3] = wmma_bf16(a1, b3, acc[1][3]);
  }
  int half = lane >> 4, nn = lane & 15;
#pragma unroll
  for (int i = 0; i < 2; ++i)
#pragma unroll
    for (int j = 0; j < 4; ++j)
#pragma unroll
      for (int r = 0; r < 8; ++r) {
        int gm = row + i * 16 + half * 8 + r;
        int gn = col + j * 16 + nn;
        C[(size_t)gm * N + gn] = (__bf16)acc[i][j][r];
      }
}

// K1v: same GEMM but stores the result transposed per (b,h):
//   vT[((b*H + h)*64 + dv) * S + s]  (so V columns are contiguous for B-fragments)
__global__ void gemm_bf16_obf16_Tv(const __bf16* __restrict__ A, const __bf16* __restrict__ Bt,
                                   __bf16* __restrict__ vT, int M, int N, int K) {
  int lane = threadIdx.x & 31, wave = threadIdx.x >> 5;
  int row = blockIdx.x * 128 + (wave >> 1) * 32;
  int col = blockIdx.y * 128 + (wave & 1) * 64;
  v8f acc[2][4];
#pragma unroll
  for (int i = 0; i < 2; ++i)
#pragma unroll
    for (int j = 0; j < 4; ++j) acc[i][j] = vzero8();

  for (int k0 = 0; k0 < K; k0 += 32) {
    v16bf a0 = load_fragA(A, K, row,      k0, lane);
    v16bf a1 = load_fragA(A, K, row + 16, k0, lane);
    v16bf b0 = load_fragB(Bt, K, col,      k0, lane);
    v16bf b1 = load_fragB(Bt, K, col + 16, k0, lane);
    v16bf b2 = load_fragB(Bt, K, col + 32, k0, lane);
    v16bf b3 = load_fragB(Bt, K, col + 48, k0, lane);
    acc[0][0] = wmma_bf16(a0, b0, acc[0][0]);
    acc[0][1] = wmma_bf16(a0, b1, acc[0][1]);
    acc[0][2] = wmma_bf16(a0, b2, acc[0][2]);
    acc[0][3] = wmma_bf16(a0, b3, acc[0][3]);
    acc[1][0] = wmma_bf16(a1, b0, acc[1][0]);
    acc[1][1] = wmma_bf16(a1, b1, acc[1][1]);
    acc[1][2] = wmma_bf16(a1, b2, acc[1][2]);
    acc[1][3] = wmma_bf16(a1, b3, acc[1][3]);
  }
  int half = lane >> 4, nn = lane & 15;
#pragma unroll
  for (int i = 0; i < 2; ++i)
#pragma unroll
    for (int j = 0; j < 4; ++j)
#pragma unroll
      for (int r = 0; r < 8; ++r) {
        int gm = row + i * 16 + half * 8 + r;
        int gn = col + j * 16 + nn;
        int b = gm >> 13;                       // / S
        int within = gm & 8191;
        int flat = within * 512 + gn;           // flat within batch
        int h  = flat >> 19;                    // / (S*64)
        int rem = flat & ((1 << 19) - 1);
        int s  = rem >> 6;
        int dv = rem & 63;
        vT[((size_t)(b * Hc + h) * 64 + dv) * Sc + s] = (__bf16)acc[i][j][r];
      }
}

// K5: GEMM with fp32 output and the reference's (nseg,b,l)->(b,nseg,l) row permute
__global__ void gemm_bf16_of32_perm(const __bf16* __restrict__ A, const __bf16* __restrict__ Bt,
                                    float* __restrict__ C, int M, int N, int K) {
  int lane = threadIdx.x & 31, wave = threadIdx.x >> 5;
  int row = blockIdx.x * 128 + (wave >> 1) * 32;
  int col = blockIdx.y * 128 + (wave & 1) * 64;
  v8f acc[2][4];
#pragma unroll
  for (int i = 0; i < 2; ++i)
#pragma unroll
    for (int j = 0; j < 4; ++j) acc[i][j] = vzero8();

  for (int k0 = 0; k0 < K; k0 += 32) {
    v16bf a0 = load_fragA(A, K, row,      k0, lane);
    v16bf a1 = load_fragA(A, K, row + 16, k0, lane);
    v16bf b0 = load_fragB(Bt, K, col,      k0, lane);
    v16bf b1 = load_fragB(Bt, K, col + 16, k0, lane);
    v16bf b2 = load_fragB(Bt, K, col + 32, k0, lane);
    v16bf b3 = load_fragB(Bt, K, col + 48, k0, lane);
    acc[0][0] = wmma_bf16(a0, b0, acc[0][0]);
    acc[0][1] = wmma_bf16(a0, b1, acc[0][1]);
    acc[0][2] = wmma_bf16(a0, b2, acc[0][2]);
    acc[0][3] = wmma_bf16(a0, b3, acc[0][3]);
    acc[1][0] = wmma_bf16(a1, b0, acc[1][0]);
    acc[1][1] = wmma_bf16(a1, b1, acc[1][1]);
    acc[1][2] = wmma_bf16(a1, b2, acc[1][2]);
    acc[1][3] = wmma_bf16(a1, b3, acc[1][3]);
  }
  int half = lane >> 4, nn = lane & 15;
#pragma unroll
  for (int i = 0; i < 2; ++i)
#pragma unroll
    for (int j = 0; j < 4; ++j)
#pragma unroll
      for (int r = 0; r < 8; ++r) {
        int g = row + i * 16 + half * 8 + r;     // row in (nseg, b, l~) order
        int nseg = g >> 11;                      // / (B*L)
        int b    = (g >> 9) & 3;                 // / L % B
        int lt   = g & 511;
        size_t orow = ((size_t)(b * NSEGc + nseg) * Lc + lt) * Dc;
        C[orow + col + j * 16 + nn] = acc[i][j][r];
      }
}

// ---------------------------------------------------------------------------
// K2: per-(b,h,segment) outer products: memT_n[dv][dk] = sum_l v[l,dv]*sigma_k[l,dk]
//     zseg_n[dk] = sum_l sigma_k[l,dk].  V comes pre-transposed (vT), only
//     sigma_k needs an LDS transpose (elu applied on the fly).
__global__ void seg_outer(const __bf16* __restrict__ kp, const __bf16* __restrict__ vT,
                          float* __restrict__ memT, float* __restrict__ zv) {
  int nseg = blockIdx.x, bh = blockIdx.y;
  const __bf16* kseg = kp + ((size_t)bh * Sc + (size_t)nseg * Lc) * DKc;
  const __bf16* vTbh = vT + (size_t)bh * 64 * Sc;

  __shared__ __bf16 ldsKT[64 * 32];   // [dk][l]
  __shared__ float  zacc[64];

  int t = threadIdx.x, lane = t & 31, wave = t >> 5;
  int half = lane >> 4, nn = lane & 15;
  if (t < 64) zacc[t] = 0.f;

  int mt  = wave >> 1;            // dv tile 0..3
  int ntA = (wave & 1) * 2;       // dk tile base {0,2}
  v8f acc[2] = {vzero8(), vzero8()};

  for (int c = 0; c < 16; ++c) {
    __syncthreads();
    int lrow = (t >> 3);
    int cb = (t & 7) * 8;
    v8bf kvv = *(const v8bf*)(kseg + (size_t)(c * 32 + lrow) * DKc + cb);
#pragma unroll
    for (int j = 0; j < 8; ++j) {
      float sk = elu1((float)kvv[j]);
      ldsKT[(cb + j) * 32 + lrow] = (__bf16)sk;
      atomicAdd(&zacc[cb + j], sk);
    }
    __syncthreads();
    v16bf a  = load_fragA(vTbh, Sc, mt * 16, nseg * Lc + c * 32, lane);
    v16bf b0 = load_fragB(ldsKT, 32, ntA * 16, 0, lane);
    v16bf b1 = load_fragB(ldsKT, 32, (ntA + 1) * 16, 0, lane);
    acc[0] = wmma_bf16(a, b0, acc[0]);
    acc[1] = wmma_bf16(a, b1, acc[1]);
  }
  __syncthreads();

  float* mslot = memT + ((size_t)bh * NSEGc + nseg) * (DKc * DVc);
#pragma unroll
  for (int j = 0; j < 2; ++j)
#pragma unroll
    for (int r = 0; r < 8; ++r) {
      int dv = mt * 16 + half * 8 + r;
      int dk = (ntA + j) * 16 + nn;
      mslot[dv * DKc + dk] = acc[j][r];
    }
  if (t < 64) zv[((size_t)bh * NSEGc + nseg) * DKc + t] = zacc[t];
}

// K3: exclusive prefix over segments (per b,h), emitted directly in bf16 for K4.
//     mem0 = 0, z0 = 1/DK.
__global__ void prefix_scan(const float* __restrict__ memT, const float* __restrict__ zv,
                            __bf16* __restrict__ memTb, __bf16* __restrict__ zvb) {
  int bh = blockIdx.x, t = threadIdx.x;
  float run[16];
#pragma unroll
  for (int i = 0; i < 16; ++i) run[i] = 0.f;
  for (int n = 0; n < NSEGc; ++n) {
    size_t slot = ((size_t)bh * NSEGc + n) * 4096 + t * 16;
    const float* p = memT + slot;
    __bf16* q = memTb + slot;
#pragma unroll
    for (int i = 0; i < 16; ++i) { q[i] = (__bf16)run[i]; run[i] += p[i]; }
  }
  if (t < 64) {
    float rz = 1.f / 64.f;
    for (int n = 0; n < NSEGc; ++n) {
      size_t zi = ((size_t)bh * NSEGc + n) * 64 + t;
      zvb[zi] = (__bf16)rz;
      rz += zv[zi];
    }
  }
}

// ---------------------------------------------------------------------------
// K4: per-(b,h,segment) attention: flash causal softmax + memory path + beta gate
__global__ void seg_attention(const __bf16* __restrict__ qp, const __bf16* __restrict__ kp,
                              const __bf16* __restrict__ vT, const __bf16* __restrict__ memTb,
                              const __bf16* __restrict__ zvb, const float* __restrict__ betas,
                              __bf16* __restrict__ att) {
  __shared__ __bf16 Pst[8 * 16 * 64];                // per-wave P staging

  int nseg = blockIdx.x, bh = blockIdx.y;
  int b = bh >> 3, h = bh & 7;
  const __bf16* qseg = qp + ((size_t)bh * Sc + (size_t)nseg * Lc) * DKc;
  const __bf16* kseg = kp + ((size_t)bh * Sc + (size_t)nseg * Lc) * DKc;
  const __bf16* vTbh = vT + (size_t)bh * 64 * Sc;
  const __bf16* memB = memTb + ((size_t)bh * NSEGc + nseg) * 4096;
  const __bf16* zB   = zvb   + ((size_t)bh * NSEGc + nseg) * 64;

  int t = threadIdx.x, lane = t & 31, wave = t >> 5;
  int half = lane >> 4, nn = lane & 15;
  int vk0 = nseg * Lc;                               // l-offset of this segment in vT

  __bf16* Pw = Pst + wave * 16 * 64;
  size_t attbase = (((size_t)nseg * Bc + b) * Hc + h) * ((size_t)Lc * DVc);

  // beta gate depends only on (h, dv): hoist out of the strip loop
  float bgate[4];
#pragma unroll
  for (int dvt = 0; dvt < 4; ++dvt)
    bgate[dvt] = 1.f / (1.f + __expf(-betas[h * 64 + dvt * 16 + nn]));

  // z fragments are strip-invariant too
  v16bf bz0 = *(const v16bf*)(zB + half * 16);
  v16bf bz1 = *(const v16bf*)(zB + 32 + half * 16);

  for (int s = 0; s < 4; ++s) {
    int qb = wave * 64 + s * 16;          // 16-row q strip

    v16bf aq0 = load_fragA(qseg, DKc, qb, 0, lane);
    v16bf aq1 = load_fragA(qseg, DKc, qb, 32, lane);
    v16bf sq0, sq1;
#pragma unroll
    for (int e = 0; e < 16; ++e) {
      sq0[e] = (__bf16)elu1((float)aq0[e]);
      sq1[e] = (__bf16)elu1((float)aq1[e]);
    }

    // memory path: sigma_q @ mem, sigma_q @ z (mem pre-converted to bf16, B-layout)
    v8f accM[4];
#pragma unroll
    for (int dvt = 0; dvt < 4; ++dvt) {
      accM[dvt] = vzero8();
      v16bf bm0 = load_fragB(memB, 64, dvt * 16, 0, lane);
      v16bf bm1 = load_fragB(memB, 64, dvt * 16, 32, lane);
      accM[dvt] = wmma_bf16(sq0, bm0, accM[dvt]);
      accM[dvt] = wmma_bf16(sq1, bm1, accM[dvt]);
    }
    v8f accZ = vzero8();
    accZ = wmma_bf16(sq0, bz0, accZ);
    accZ = wmma_bf16(sq1, bz1, accZ);

    // flash causal softmax attention
    v8f accO[4];
#pragma unroll
    for (int dvt = 0; dvt < 4; ++dvt) accO[dvt] = vzero8();
    float rowmax[8], rowsum[8];
#pragma unroll
    for (int r = 0; r < 8; ++r) { rowmax[r] = NEG_BIG; rowsum[r] = 0.f; }

    int cmax = (qb + 15) >> 6;
    for (int c = 0; c <= cmax; ++c) {
      int j0 = c * 64;
      v8f sc[4];
#pragma unroll
      for (int jt = 0; jt < 4; ++jt) {
        v16bf bk0 = load_fragB(kseg, DKc, j0 + jt * 16, 0, lane);
        v16bf bk1 = load_fragB(kseg, DKc, j0 + jt * 16, 32, lane);
        v8f scc = vzero8();
        scc = wmma_bf16(aq0, bk0, scc);
        scc = wmma_bf16(aq1, bk1, scc);
        sc[jt] = scc;
      }
      float cmx[8];
#pragma unroll
      for (int r = 0; r < 8; ++r) cmx[r] = NEG_BIG;
#pragma unroll
      for (int jt = 0; jt < 4; ++jt)
#pragma unroll
        for (int r = 0; r < 8; ++r) {
          int gi = qb + half * 8 + r;
          int gj = j0 + jt * 16 + nn;
          float sv = (gj <= gi) ? sc[jt][r] * 0.125f : NEG_BIG;
          sc[jt][r] = sv;
          cmx[r] = fmaxf(cmx[r], sv);
        }
#pragma unroll
      for (int r = 0; r < 8; ++r) {
#pragma unroll
        for (int m = 1; m < 16; m <<= 1) cmx[r] = fmaxf(cmx[r], __shfl_xor(cmx[r], m, 32));
        float nm = fmaxf(rowmax[r], cmx[r]);
        float corr = __expf(rowmax[r] - nm);
        rowsum[r] *= corr;
        rowmax[r] = nm;
#pragma unroll
        for (int dvt = 0; dvt < 4; ++dvt) accO[dvt][r] *= corr;
      }
      float psum[8];
#pragma unroll
      for (int r = 0; r < 8; ++r) psum[r] = 0.f;
#pragma unroll
      for (int jt = 0; jt < 4; ++jt)
#pragma unroll
        for (int r = 0; r < 8; ++r) {
          float pv = __expf(sc[jt][r] - rowmax[r]);
          psum[r] += pv;
          Pw[(half * 8 + r) * 64 + jt * 16 + nn] = (__bf16)pv;
        }
#pragma unroll
      for (int r = 0; r < 8; ++r) {
#pragma unroll
        for (int m = 1; m < 16; m <<= 1) psum[r] += __shfl_xor(psum[r], m, 32);
        rowsum[r] += psum[r];
      }
      v16bf pa0 = load_fragA(Pw, 64, 0, 0, lane);
      v16bf pa1 = load_fragA(Pw, 64, 0, 32, lane);
#pragma unroll
      for (int dvt = 0; dvt < 4; ++dvt) {
        v16bf bv0 = load_fragB(vTbh, Sc, dvt * 16, vk0 + j0, lane);
        v16bf bv1 = load_fragB(vTbh, Sc, dvt * 16, vk0 + j0 + 32, lane);
        accO[dvt] = wmma_bf16(pa0, bv0, accO[dvt]);
        accO[dvt] = wmma_bf16(pa1, bv1, accO[dvt]);
      }
    }

    // combine: beta * att_mem + (1-beta) * att_dot, store bf16
#pragma unroll
    for (int dvt = 0; dvt < 4; ++dvt)
#pragma unroll
      for (int r = 0; r < 8; ++r) {
        int gi = qb + half * 8 + r;
        int dv = dvt * 16 + nn;
        float att_dot = accO[dvt][r] / rowsum[r];
        float att_mem = accM[dvt][r] / accZ[r];
        att[attbase + (size_t)gi * DVc + dv] =
            (__bf16)(bgate[dvt] * att_mem + (1.f - bgate[dvt]) * att_dot);
      }
  }
}

// ---------------------------------------------------------------------------
extern "C" void kernel_launch(void* const* d_in, const int* in_sizes, int n_in,
                              void* d_out, int out_size, void* d_ws, size_t ws_size,
                              hipStream_t stream) {
  (void)in_sizes; (void)n_in; (void)out_size; (void)ws_size;
  const float* x     = (const float*)d_in[0];
  const float* Wk    = (const float*)d_in[1];
  const float* Wv    = (const float*)d_in[2];
  const float* Wq    = (const float*)d_in[3];
  const float* betas = (const float*)d_in[4];
  const float* Wout  = (const float*)d_in[5];
  float* out = (float*)d_out;

  char* p = (char*)d_ws;
  auto alloc = [&](size_t bytes) -> char* {
    char* r = p;
    p += (bytes + 255) & ~(size_t)255;
    return r;
  };
  const size_t NBIG = (size_t)MROWS * Dc;                 // 16.78M elems
  __bf16* xb    = (__bf16*)alloc(NBIG * 2);
  __bf16* wkT   = (__bf16*)alloc((size_t)512 * 512 * 2);
  __bf16* wvT   = (__bf16*)alloc((size_t)512 * 512 * 2);
  __bf16* wqT   = (__bf16*)alloc((size_t)512 * 512 * 2);
  __bf16* woT   = (__bf16*)alloc((size_t)512 * 512 * 2);
  __bf16* kp    = (__bf16*)alloc(NBIG * 2);
  __bf16* vT    = (__bf16*)alloc(NBIG * 2);               // V pre-transposed per (b,h)
  __bf16* qp    = (__bf16*)alloc(NBIG * 2);
  float*  memT  = (float*)alloc((size_t)BHc * NSEGc * 4096 * 4);
  float*  zv    = (float*)alloc((size_t)BHc * NSEGc * 64 * 4);
  __bf16* memTb = (__bf16*)alloc((size_t)BHc * NSEGc * 4096 * 2);
  __bf16* zvb   = (__bf16*)alloc((size_t)BHc * NSEGc * 64 * 2);
  __bf16* att   = (__bf16*)alloc(NBIG * 2);

  // 0) convert inputs to bf16 (weights transposed so B-fragments load contiguously)
  cvt_f32_bf16<<<dim3(4096), dim3(256), 0, stream>>>(x, xb, (int)NBIG);
  cvtT512_f32_bf16<<<dim3(1024), dim3(256), 0, stream>>>(Wk, wkT);
  cvtT512_f32_bf16<<<dim3(1024), dim3(256), 0, stream>>>(Wv, wvT);
  cvtT512_f32_bf16<<<dim3(1024), dim3(256), 0, stream>>>(Wq, wqT);
  cvtT512_f32_bf16<<<dim3(1024), dim3(256), 0, stream>>>(Wout, woT);

  // 1) projections: k, q row-major; v stored transposed per (b,h)
  dim3 ggrid(MROWS / 128, Dc / 128);
  gemm_bf16_obf16<<<ggrid, dim3(256), 0, stream>>>(xb, wkT, kp, MROWS, Dc, Dc);
  gemm_bf16_obf16_Tv<<<ggrid, dim3(256), 0, stream>>>(xb, wvT, vT, MROWS, Dc, Dc);
  gemm_bf16_obf16<<<ggrid, dim3(256), 0, stream>>>(xb, wqT, qp, MROWS, Dc, Dc);

  // 2) per-segment outer products  3) exclusive prefix over segments (bf16 out)
  seg_outer<<<dim3(NSEGc, BHc), dim3(256), 0, stream>>>(kp, vT, memT, zv);
  prefix_scan<<<dim3(BHc), dim3(256), 0, stream>>>(memT, zv, memTb, zvb);

  // 4) per-segment attention (flash softmax + linear-memory path + beta gate)
  seg_attention<<<dim3(NSEGc, BHc), dim3(256), 0, stream>>>(qp, kp, vT, memTb, zvb, betas, att);

  // 5) output projection with row permutation (nseg,b,l) -> (b, nseg*L+l)
  gemm_bf16_of32_perm<<<ggrid, dim3(256), 0, stream>>>(att, woT, out, MROWS, Dc, Dc);
}
